// PMA_10814727651836
// MI455X (gfx1250) — matrix operand order repaired and checked by reading, not compiled, not run on hardware
//
#include <hip/hip_runtime.h>
#include <hip/hip_bf16.h>
#include <stdint.h>

#define BATCH   16
#define NSEQ    8192
#define DIM     512
#define HEADS   8
#define HD      64
#define KQ      4
#define SPLITS  16
#define ROWS_PER_SPLIT (NSEQ / SPLITS)   // 512
#define CHUNKS  (ROWS_PER_SPLIT / 16)    // 32
#define LDS_ROW_U 268                    // uints per LDS row (bank-conflict-free padding)
#define PACKW_U (DIM * DIM / 2)          // 131072 uints per packed weight matrix

typedef __attribute__((ext_vector_type(16))) __bf16        v16bf;
typedef __attribute__((ext_vector_type(8)))  float         v8f;
typedef __attribute__((ext_vector_type(8)))  unsigned int  v8u;

__device__ __forceinline__ unsigned int bf16rne(float f) {
  unsigned int u = __float_as_uint(f);
  u += 0x7FFFu + ((u >> 16) & 1u);          // round-to-nearest-even
  return u >> 16;
}
__device__ __forceinline__ unsigned int pack2(float lo, float hi) {
  return bf16rne(lo) | (bf16rne(hi) << 16);
}
__device__ __forceinline__ v16bf make_op(uint4 a, uint4 b) {
  v8u u;
  u[0] = a.x; u[1] = a.y; u[2] = a.z; u[3] = a.w;
  u[4] = b.x; u[5] = b.y; u[6] = b.z; u[7] = b.w;
  return __builtin_bit_cast(v16bf, u);
}

// ---------------------------------------------------------------------------
// Pack 512x512 fp32 weight (torch Linear layout: out x in) into bf16 WMMA
// B-operand tiles for C = Xc(16xK) @ W^T. Tile (nblk, kblk): lane l -> col
// n = nblk*16 + (l&15); K base = kblk*32 + (l>=16 ? 16 : 0); dword j holds
// K = base+2j (lo) / base+2j+1 (hi). Lane's 8 dwords are contiguous (2x b128).
// ---------------------------------------------------------------------------
__global__ void pma_pack_weight(const float* __restrict__ W,
                                unsigned int* __restrict__ out) {
  int idx  = blockIdx.x * 256 + threadIdx.x;   // 0..131071
  int tile = idx >> 8;
  int r    = idx & 255;
  int lane = r >> 3;
  int j    = r & 7;
  int nblk = tile >> 4;
  int kblk = tile & 15;
  int n    = nblk * 16 + (lane & 15);
  int k0   = kblk * 32 + (lane & 16) + 2 * j;
  out[idx] = pack2(W[(size_t)n * DIM + k0], W[(size_t)n * DIM + k0 + 1]);
}

// q = 0.125 * (seed @ Wq^T + bq)   (batch-invariant; folds the 1/sqrt(64))
__global__ void pma_qproj(const float* __restrict__ seed,
                          const float* __restrict__ Wq,
                          const float* __restrict__ bq,
                          float* __restrict__ qbuf) {
  for (int o = threadIdx.x; o < KQ * DIM; o += blockDim.x) {
    int qi = o >> 9;
    int d  = o & 511;
    float s = bq[d];
    const float* sr = seed + qi * DIM;
    const float* wr = Wq + (size_t)d * DIM;
    for (int j = 0; j < DIM; ++j) s += sr[j] * wr[j];
    qbuf[o] = 0.125f * s;
  }
}

// ---------------------------------------------------------------------------
// Fused streaming attention. Block = (split, b), 256 threads = 8 waves; wave
// w owns head w.  Double-buffered LDS: global loads for chunk ch+1 are issued
// before the WMMA loop of chunk ch; convert+ds_store happen after it.  The
// kb loop computes K and V tiles together from one A operand (8 WMMAs / kb).
// Wv tiles sit at a constant +512KB offset from Wk tiles (single base addr).
// ---------------------------------------------------------------------------
__global__ void __launch_bounds__(256) pma_attn_main(
    const float* __restrict__ X,
    const unsigned int* __restrict__ packW,   // [packWk | packWv] contiguous
    const float* __restrict__ qbuf,
    const float* __restrict__ bk,
    const float* __restrict__ bv,
    float* __restrict__ pbuf) {
  __shared__ __align__(16) unsigned int ldsX[2][16 * LDS_ROW_U];

  const int tid   = threadIdx.x;
  const int w     = tid >> 5;   // wave id == head
  const int lane  = tid & 31;
  const int lm    = lane & 15;  // A: row M / B,C: column within tile
  const int g     = lane >> 4;  // half-wave group
  const int b     = blockIdx.y;
  const int split = blockIdx.x;

  float qv[KQ][4];
  float bkv[4], bvv[4];
#pragma unroll
  for (int t = 0; t < 4; ++t) {
    int c = w * HD + t * 16 + lm;
    bkv[t] = bk[c];
    bvv[t] = bv[c];
#pragma unroll
    for (int qi = 0; qi < KQ; ++qi) qv[qi][t] = qbuf[qi * DIM + c];
  }

  float mrun[KQ], lrun[KQ], oacc[KQ][4];
#pragma unroll
  for (int qi = 0; qi < KQ; ++qi) {
    mrun[qi] = -3.0e38f;
    lrun[qi] = 0.0f;
#pragma unroll
    for (int t = 0; t < 4; ++t) oacc[qi][t] = 0.0f;
  }

  const int stg_row = tid >> 4;
  const int stg_seg = tid & 15;

  auto load_chunk = [&](int ch, float4 (&reg)[8]) {
    const int n0 = split * ROWS_PER_SPLIT + ch * 16;
    const float4* src =
        (const float4*)(X + ((size_t)b * NSEQ + (n0 + stg_row)) * DIM + stg_seg * 32);
#pragma unroll
    for (int i = 0; i < 8; ++i) reg[i] = src[i];
  };
  auto store_chunk = [&](int buf, const float4 (&reg)[8]) {
    unsigned int* dst = &ldsX[buf][stg_row * LDS_ROW_U + stg_seg * 16];
#pragma unroll
    for (int i = 0; i < 8; ++i) {
      dst[i * 2]     = pack2(reg[i].x, reg[i].y);
      dst[i * 2 + 1] = pack2(reg[i].z, reg[i].w);
    }
  };

  float4 sreg[8];
  load_chunk(0, sreg);
  store_chunk(0, sreg);
  __syncthreads();

  for (int ch = 0; ch < CHUNKS; ++ch) {
    const int  cur  = ch & 1;
    const bool more = (ch + 1) < CHUNKS;

    // issue next chunk's global loads before the compute (latency hidden
    // behind the WMMA loop; convert+store happen after compute)
    if (more) load_chunk(ch + 1, sreg);

    // ---- fused K/V GEMM: 8 WMMAs per kb step from one A operand ----
    v8f accK[4], accV[4];
#pragma unroll
    for (int t = 0; t < 4; ++t) { v8f z = {}; accK[t] = z; accV[t] = z; }
    for (int kb = 0; kb < 16; ++kb) {
      const uint4* ap = (const uint4*)&ldsX[cur][lm * LDS_ROW_U + kb * 16 + g * 8];
      v16bf A = make_op(ap[0], ap[1]);
      const unsigned int* wb = packW + ((w * 64 + kb) * 256 + lane * 8);
#pragma unroll
      for (int t = 0; t < 4; ++t) {
        const uint4* bpk = (const uint4*)(wb + t * 4096);
        v16bf Bk = make_op(bpk[0], bpk[1]);
        accK[t] = __builtin_amdgcn_wmma_f32_16x16x32_bf16(
            false, A, false, Bk, (short)0, accK[t], false, false);
        const uint4* bpv = (const uint4*)(wb + PACKW_U + t * 4096);
        v16bf Bv = make_op(bpv[0], bpv[1]);
        accV[t] = __builtin_amdgcn_wmma_f32_16x16x32_bf16(
            false, A, false, Bv, (short)0, accV[t], false, false);
      }
    }

    // ---- scores + online softmax ----
    // C layout: lane holds column c_t = w*64+t*16+lm, rows m = r + 8*g.
    float e_[KQ][8];
#pragma unroll
    for (int qi = 0; qi < KQ; ++qi) {
      float p[8];
#pragma unroll
      for (int r = 0; r < 8; ++r) {
        p[r] = qv[qi][0] * (accK[0][r] + bkv[0])
             + qv[qi][1] * (accK[1][r] + bkv[1])
             + qv[qi][2] * (accK[2][r] + bkv[2])
             + qv[qi][3] * (accK[3][r] + bkv[3]);
      }
#pragma unroll
      for (int msk = 1; msk <= 8; msk <<= 1)
#pragma unroll
        for (int r = 0; r < 8; ++r) p[r] += __shfl_xor(p[r], msk, 32);
      // p[r] == s[qi][r+8g], replicated across the 16 lanes of each half
      float mx = p[0];
#pragma unroll
      for (int r = 1; r < 8; ++r) mx = fmaxf(mx, p[r]);
      mx = fmaxf(mx, __shfl_xor(mx, 16, 32));
      float mnew  = fmaxf(mrun[qi], mx);
      float alpha = __expf(mrun[qi] - mnew);
      float ps = 0.0f;
#pragma unroll
      for (int r = 0; r < 8; ++r) { e_[qi][r] = __expf(p[r] - mnew); ps += e_[qi][r]; }
      ps += __shfl_xor(ps, 16, 32);
      lrun[qi] = lrun[qi] * alpha + ps;
      mrun[qi] = mnew;
#pragma unroll
      for (int t = 0; t < 4; ++t) oacc[qi][t] *= alpha;
    }

    // ---- o += sum_m e[m] * (V[m][c]+bv) (cross-half combine, 16 rows) ----
#pragma unroll
    for (int qi = 0; qi < KQ; ++qi)
#pragma unroll
      for (int t = 0; t < 4; ++t) {
        float part = 0.0f;
#pragma unroll
        for (int r = 0; r < 8; ++r) part += e_[qi][r] * (accV[t][r] + bvv[t]);
        part += __shfl_xor(part, 16, 32);
        oacc[qi][t] += part;
      }

    if (more) store_chunk(1 - cur, sreg);
    __syncthreads();
  }

  // ---- store split partials: {m, l, o[64]} ----
  float* pb = pbuf + ((((size_t)b * SPLITS + split) * HEADS + w) * KQ) * 66;
#pragma unroll
  for (int qi = 0; qi < KQ; ++qi) {
    if (lane == 0) { pb[qi * 66 + 0] = mrun[qi]; pb[qi * 66 + 1] = lrun[qi]; }
    if (lane < 16) {
#pragma unroll
      for (int t = 0; t < 4; ++t) pb[qi * 66 + 2 + t * 16 + lm] = oacc[qi][t];
    }
  }
}

// Merge per-split softmax partials -> attn output rows (B, KQ, DIM)
__global__ void pma_combine(const float* __restrict__ pbuf,
                            float* __restrict__ attnbuf) {
  int b = blockIdx.x >> 3;
  int h = blockIdx.x & 7;
  int c = threadIdx.x;   // 0..63
  for (int qi = 0; qi < KQ; ++qi) {
    float M = -3.0e38f;
    for (int s = 0; s < SPLITS; ++s) {
      const float* pb = pbuf + ((((size_t)b * SPLITS + s) * HEADS + h) * KQ + qi) * 66;
      M = fmaxf(M, pb[0]);
    }
    float L = 0.0f, o = 0.0f;
    for (int s = 0; s < SPLITS; ++s) {
      const float* pb = pbuf + ((((size_t)b * SPLITS + s) * HEADS + h) * KQ + qi) * 66;
      float sc = __expf(pb[0] - M);
      L += pb[1] * sc;
      o += sc * pb[2 + c];
    }
    attnbuf[(size_t)(b * KQ + qi) * DIM + h * HD + c] = o / L;
  }
}

// out = LayerNorm(attn @ Wo^T + bo + seed) * gamma + beta
__global__ void pma_outln(const float* __restrict__ attnbuf,
                          const float* __restrict__ Wo,
                          const float* __restrict__ bo,
                          const float* __restrict__ seed,
                          const float* __restrict__ gamma,
                          const float* __restrict__ beta,
                          float* __restrict__ out) {
  __shared__ float row[DIM];
  __shared__ float red[DIM];
  int b  = blockIdx.x >> 2;
  int qi = blockIdx.x & 3;
  int d  = threadIdx.x;           // 0..511
  row[d] = attnbuf[(size_t)(b * KQ + qi) * DIM + d];
  __syncthreads();
  float acc = bo[d];
  const float* wr = Wo + (size_t)d * DIM;
  for (int j = 0; j < DIM; ++j) acc += row[j] * wr[j];
  float hv = acc + seed[qi * DIM + d];
  red[d] = hv;
  __syncthreads();
  for (int s = DIM / 2; s > 0; s >>= 1) {
    if (d < s) red[d] += red[d + s];
    __syncthreads();
  }
  float mu = red[0] * (1.0f / DIM);
  __syncthreads();
  float diff = hv - mu;
  red[d] = diff * diff;
  __syncthreads();
  for (int s = DIM / 2; s > 0; s >>= 1) {
    if (d < s) red[d] += red[d + s];
    __syncthreads();
  }
  float var = red[0] * (1.0f / DIM);
  out[(size_t)(b * KQ + qi) * DIM + d] = diff * rsqrtf(var + 1e-6f) * gamma[d] + beta[d];
}

extern "C" void kernel_launch(void* const* d_in, const int* in_sizes, int n_in,
                              void* d_out, int out_size, void* d_ws, size_t ws_size,
                              hipStream_t stream) {
  const float* X     = (const float*)d_in[0];
  const float* seed  = (const float*)d_in[1];
  const float* Wq    = (const float*)d_in[2];
  const float* bq    = (const float*)d_in[3];
  const float* Wk    = (const float*)d_in[4];
  const float* bk    = (const float*)d_in[5];
  const float* Wv    = (const float*)d_in[6];
  const float* bv    = (const float*)d_in[7];
  const float* Wo    = (const float*)d_in[8];
  const float* bo    = (const float*)d_in[9];
  const float* gamma = (const float*)d_in[10];
  const float* beta  = (const float*)d_in[11];
  float* out = (float*)d_out;

  // workspace layout (bytes): packWk 512K | packWv 512K | qbuf 8K | pbuf ~2.06M | attnbuf 128K
  char* ws = (char*)d_ws;
  unsigned int* packWk = (unsigned int*)(ws);
  unsigned int* packWv = (unsigned int*)(ws + 524288);   // contiguous after packWk
  float* qbuf    = (float*)(ws + 1048576);
  float* pbuf    = (float*)(ws + 1048576 + 16384);
  float* attnbuf = (float*)(ws + 1048576 + 16384 + 2162688);

  pma_pack_weight<<<512, 256, 0, stream>>>(Wk, packWk);
  pma_pack_weight<<<512, 256, 0, stream>>>(Wv, packWv);
  pma_qproj<<<1, 256, 0, stream>>>(seed, Wq, bq, qbuf);
  pma_attn_main<<<dim3(SPLITS, BATCH), 256, 0, stream>>>(X, packWk, qbuf, bk, bv, pbuf);
  pma_combine<<<BATCH * HEADS, HD, 0, stream>>>(pbuf, attnbuf);
  pma_outln<<<BATCH * KQ, DIM, 0, stream>>>(attnbuf, Wo, bo, seed, gamma, beta, out);

  (void)in_sizes; (void)n_in; (void)out_size; (void)ws_size;
}
